// ManifoldAttentionNoAttn_84670985273586
// MI455X (gfx1250) — compile-verified, matrix operand order checked by reading, not run on hardware
//
#include <hip/hip_runtime.h>
#include <hip/hip_bf16.h>

// ---------------------------------------------------------------------------
// ManifoldAttentionNoAttn for MI455X (gfx1250), wave32 + WMMA bf16.
// GEMMs: v_wmma_f32_16x16x32_bf16, fp32 accumulate, 2x4 / 1x4 / 4x2 tile
// blocking per wave. The block-shared operand (B slab, or A for the
// covariance GEMM) is staged through LDS in 33KB chunks so the 8 waves of a
// workgroup read it via ds_load_b128 instead of 8x-replicated L2 traffic.
// Rows are padded +8 bf16 to spread the 16 fragment rows across LDS banks.
// ---------------------------------------------------------------------------

typedef __attribute__((ext_vector_type(16))) __bf16 v16bf;
typedef __attribute__((ext_vector_type(8)))  __bf16 v8bf;
typedef __attribute__((ext_vector_type(8)))  float  v8f;

#define B_   4
#define T_   2048
#define D_   1024
#define R_   64
#define BT_  (B_*T_)
#define KC_  256          // LDS staging chunk (K elements)
#define LDP_ (KC_ + 8)    // padded LDS row length
#define EPSF 1e-5f
#define LAMF 0.05f
#define RHOF 0.9f

__device__ __forceinline__ float gelu_exact(float v) {
  return 0.5f * v * (1.0f + erff(v * 0.70710678118654752f));
}

// Fragment from a (rows, K) row-major bf16 matrix in GLOBAL memory.
__device__ __forceinline__ v16bf frag_row(const __bf16* __restrict__ p,
                                          int row, int ld, int k0, int lane) {
  const int kb = (lane >> 4) << 3;                 // 0 or 8
  const __bf16* b = p + (long)row * ld + k0 + kb;
  v8bf lo = *(const v8bf*)(b);                     // K = kb .. kb+7
  v8bf hi = *(const v8bf*)(b + 16);                // K = 16+kb .. 16+kb+7
  v16bf r;
#pragma unroll
  for (int i = 0; i < 8; ++i) { r[i] = lo[i]; r[i + 8] = hi[i]; }
  return r;
}

// Fragment from an LDS row (pointer to start of the row).
__device__ __forceinline__ v16bf frag_s(const __bf16* rowp, int k0, int lane) {
  const int kb = (lane >> 4) << 3;
  const __bf16* b = rowp + k0 + kb;
  v8bf lo = *(const v8bf*)(b);
  v8bf hi = *(const v8bf*)(b + 16);
  v16bf r;
#pragma unroll
  for (int i = 0; i < 8; ++i) { r[i] = lo[i]; r[i + 8] = hi[i]; }
  return r;
}

#define WMMA_BF16(a, b, c) \
  __builtin_amdgcn_wmma_f32_16x16x32_bf16(false, (a), false, (b), (short)0, (c), false, false)

// ---------------------------------------------------------------------------
__global__ void k_f32_to_bf16(const float* __restrict__ src,
                              __bf16* __restrict__ dst, int n) {
  for (int i = blockIdx.x * blockDim.x + threadIdx.x; i < n;
       i += gridDim.x * blockDim.x)
    dst[i] = (__bf16)src[i];
}

// z = gelu(x @ w_in^T)   M=BT, N=R(=64, fully blocked), K=D. 16x64 per wave.
// B (w_in, 64x1024) staged in LDS chunks, shared by all 8 waves.
__global__ void k_gemm_z(const __bf16* __restrict__ xb,
                         const __bf16* __restrict__ wi,
                         __bf16* __restrict__ z) {
  __shared__ __bf16 Bs[64][LDP_];
  const int tid = threadIdx.x;
  const int wave = blockIdx.x * 8 + (tid >> 5);   // 512 m-tiles
  const int lane = tid & 31;
  const int ar = wave * 16 + (lane & 15);
  v8f acc[4] = {};
  for (int kc = 0; kc < D_; kc += KC_) {
    __syncthreads();
    for (int idx = tid * 8; idx < 64 * KC_; idx += 2048) {
      const int rr = idx >> 8, kk = idx & (KC_ - 1);
      *(v8bf*)&Bs[rr][kk] = *(const v8bf*)(wi + (long)rr * D_ + kc + kk);
    }
    __syncthreads();
    for (int k0 = 0; k0 < KC_; k0 += 32) {
      v16bf a = frag_row(xb, ar, D_, kc + k0, lane);
      v16bf bf[4];
#pragma unroll
      for (int j = 0; j < 4; ++j)
        bf[j] = frag_s(&Bs[j * 16 + (lane & 15)][0], k0, lane);
#pragma unroll
      for (int j = 0; j < 4; ++j) acc[j] = WMMA_BF16(a, bf[j], acc[j]);
    }
  }
  const int rbase = wave * 16 + ((lane >> 4) << 3);
  const int col = lane & 15;
#pragma unroll
  for (int j = 0; j < 4; ++j)
#pragma unroll
    for (int r = 0; r < 8; ++r)
      z[(long)(rbase + r) * R_ + col + j * 16] = (__bf16)gelu_exact(acc[j][r]);
}

// s = z @ w_out_s^T + b_s ; xp = x - anchor + s.  M=BT, N=D, K=R.
// 32x64 per wave; block shares one 64x64 B slab (8KB, staged once).
__global__ void k_gemm_s_xp(const __bf16* __restrict__ z,
                            const __bf16* __restrict__ wos,
                            const float* __restrict__ bs,
                            const float* __restrict__ x,
                            float* __restrict__ xp,
                            __bf16* __restrict__ xpb,
                            __bf16* __restrict__ xpT) {
  __shared__ __bf16 Bs[64][R_ + 8];
  const int tid = threadIdx.x;
  const int wave = blockIdx.x * 8 + (tid >> 5);
  const int lane = tid & 31;
  const int bn = wave >> 8, bm = wave & 255;  // block shares bn
  const int m0 = bm * 32, n0 = bn * 64;
  for (int idx = tid * 8; idx < 64 * R_; idx += 2048) {
    const int rr = idx >> 6, kk = idx & 63;
    *(v8bf*)&Bs[rr][kk] = *(const v8bf*)(wos + (long)(n0 + rr) * R_ + kk);
  }
  __syncthreads();
  v8f acc[2][4] = {};
#pragma unroll
  for (int k0 = 0; k0 < R_; k0 += 32) {
    v16bf a[2], bf[4];
#pragma unroll
    for (int i = 0; i < 2; ++i)
      a[i] = frag_row(z, m0 + i * 16 + (lane & 15), R_, k0, lane);
#pragma unroll
    for (int j = 0; j < 4; ++j)
      bf[j] = frag_s(&Bs[j * 16 + (lane & 15)][0], k0, lane);
#pragma unroll
    for (int i = 0; i < 2; ++i)
#pragma unroll
      for (int j = 0; j < 4; ++j)
        acc[i][j] = WMMA_BF16(a[i], bf[j], acc[i][j]);
  }
#pragma unroll
  for (int i = 0; i < 2; ++i) {
    const int rbase = m0 + i * 16 + ((lane >> 4) << 3);
    const int b = rbase >> 11;           // T_ = 2048
    const int t0 = rbase & (T_ - 1);
#pragma unroll
    for (int j = 0; j < 4; ++j) {
      const int col = n0 + j * 16 + (lane & 15);
      const float bias = bs[col];
      v8bf pk;
#pragma unroll
      for (int r = 0; r < 8; ++r) {
        const long idx = (long)(rbase + r) * D_ + col;
        const float v = x[idx] - ((col == 0) ? 1.0f : 0.0f) + acc[i][j][r] + bias;
        xp[idx] = v;
        xpb[idx] = (__bf16)v;
        pk[r] = (__bf16)v;
      }
      *(v8bf*)(xpT + (long)b * D_ * T_ + (long)col * T_ + t0) = pk;
    }
  }
}

// V0 = identity columns (orthonorm(E) == E)
__global__ void k_init_V(float* __restrict__ Vt, __bf16* __restrict__ Vtb) {
  const int i = blockIdx.x * blockDim.x + threadIdx.x;
  if (i >= B_ * R_ * D_) return;
  const int d = i % D_;
  const int r = (i / D_) % R_;
  const float v = (d == r) ? 1.0f : 0.0f;
  Vt[i] = v;
  Vtb[i] = (__bf16)v;
}

// Yt[b][r][t] = (xp @ V)[t][r]. Per batch: M=T, N=R (fully blocked), K=D.
// B (Vt slab, 64x1024) staged in LDS chunks; packed transposed stores.
__global__ void k_gemm_Yt(const __bf16* __restrict__ xpb,
                          const __bf16* __restrict__ Vtb,
                          __bf16* __restrict__ Yt) {
  __shared__ __bf16 Bs[64][LDP_];
  const int tid = threadIdx.x;
  const int wave = blockIdx.x * 8 + (tid >> 5);
  const int lane = tid & 31;
  const int b = wave >> 7;             // same batch for whole block (8|128)
  const int tm = wave & 127;
  const __bf16* A  = xpb + (long)b * T_ * D_;
  const __bf16* Bp = Vtb + (long)b * R_ * D_;
  const int ar = tm * 16 + (lane & 15);
  v8f acc[4] = {};
  for (int kc = 0; kc < D_; kc += KC_) {
    __syncthreads();
    for (int idx = tid * 8; idx < 64 * KC_; idx += 2048) {
      const int rr = idx >> 8, kk = idx & (KC_ - 1);
      *(v8bf*)&Bs[rr][kk] = *(const v8bf*)(Bp + (long)rr * D_ + kc + kk);
    }
    __syncthreads();
    for (int k0 = 0; k0 < KC_; k0 += 32) {
      v16bf a = frag_row(A, ar, D_, kc + k0, lane);
      v16bf bf[4];
#pragma unroll
      for (int j = 0; j < 4; ++j)
        bf[j] = frag_s(&Bs[j * 16 + (lane & 15)][0], k0, lane);
#pragma unroll
      for (int j = 0; j < 4; ++j) acc[j] = WMMA_BF16(a, bf[j], acc[j]);
    }
  }
  __bf16* O = Yt + (long)b * R_ * T_;
  const int t0 = tm * 16 + ((lane >> 4) << 3);
#pragma unroll
  for (int j = 0; j < 4; ++j) {
    const int col = j * 16 + (lane & 15);
    v8bf pk;
#pragma unroll
    for (int r = 0; r < 8; ++r) pk[r] = (__bf16)acc[j][r];
    *(v8bf*)(O + (long)col * T_ + t0) = pk;
  }
}

// Wt[b][r][d] = (xp^T @ Y)[d][r]/T + EPS*Vt. Per batch: M=R(=64, fully
// blocked), N=D, K=T. 64x32 per wave. A (Yt, block-shared) staged in LDS.
__global__ void k_gemm_Wt(const __bf16* __restrict__ Yt,
                          const __bf16* __restrict__ xpT,
                          const float* __restrict__ Vt,
                          float* __restrict__ Wt,
                          __bf16* __restrict__ Wtb) {
  __shared__ __bf16 As[64][LDP_];
  const int tid = threadIdx.x;
  const int wave = blockIdx.x * 8 + (tid >> 5);
  const int lane = tid & 31;
  const int b = wave >> 5;             // same batch for whole block (8|32)
  const int n0 = (wave & 31) * 32;
  const __bf16* Ap = Yt  + (long)b * R_ * T_;
  const __bf16* Bp = xpT + (long)b * D_ * T_;
  v8f acc[4][2] = {};
  for (int kc = 0; kc < T_; kc += KC_) {
    __syncthreads();
    for (int idx = tid * 8; idx < 64 * KC_; idx += 2048) {
      const int rr = idx >> 8, kk = idx & (KC_ - 1);
      *(v8bf*)&As[rr][kk] = *(const v8bf*)(Ap + (long)rr * T_ + kc + kk);
    }
    __syncthreads();
    for (int k0 = 0; k0 < KC_; k0 += 32) {
      v16bf a[4], bf[2];
#pragma unroll
      for (int i = 0; i < 4; ++i)
        a[i] = frag_s(&As[i * 16 + (lane & 15)][0], k0, lane);
#pragma unroll
      for (int j = 0; j < 2; ++j)
        bf[j] = frag_row(Bp, n0 + j * 16 + (lane & 15), T_, kc + k0, lane);
#pragma unroll
      for (int i = 0; i < 4; ++i)
#pragma unroll
        for (int j = 0; j < 2; ++j)
          acc[i][j] = WMMA_BF16(a[i], bf[j], acc[i][j]);
    }
  }
  const float invT = 1.0f / (float)T_;
  const long base = (long)b * R_ * D_;
#pragma unroll
  for (int i = 0; i < 4; ++i) {
    const int rbase = i * 16 + ((lane >> 4) << 3);
#pragma unroll
    for (int j = 0; j < 2; ++j) {
      const int col = n0 + j * 16 + (lane & 15);
#pragma unroll
      for (int r = 0; r < 8; ++r) {
        const long idx = base + (long)(rbase + r) * D_ + col;
        const float wv = acc[i][j][r] * invT + EPSF * Vt[idx];
        Wt[idx] = wv;
        Wtb[idx] = (__bf16)wv;
      }
    }
  }
}

// G = Wt @ Wt^T (64x64 Gram, per batch)   M=N=R, K=D
__global__ void k_gemm_G(const __bf16* __restrict__ Wtb,
                         float* __restrict__ G) {
  const int wave = blockIdx.x * 8 + (threadIdx.x >> 5);
  const int lane = threadIdx.x & 31;
  const int b = wave >> 4;
  const int w = wave & 15;
  const int tn = w & 3, tm = w >> 2;
  const __bf16* A = Wtb + (long)b * R_ * D_;
  const int ar = tm * 16 + (lane & 15);
  const int br = tn * 16 + (lane & 15);
  v8f acc = {};
#pragma unroll 2
  for (int k0 = 0; k0 < D_; k0 += 32) {
    v16bf a = frag_row(A, ar, D_, k0, lane);
    v16bf bb = frag_row(A, br, D_, k0, lane);
    acc = WMMA_BF16(a, bb, acc);
  }
  const int rbase = tm * 16 + ((lane >> 4) << 3);
  const int col = tn * 16 + (lane & 15);
#pragma unroll
  for (int r = 0; r < 8; ++r)
    G[(long)b * R_ * R_ + (long)(rbase + r) * R_ + col] = acc[r];
}

// Cholesky G = L L^T, 64x64, one block (64 threads) per batch
__global__ void k_cholesky(const float* __restrict__ G,
                           float* __restrict__ Lg) {
  __shared__ float A[R_][R_];
  const int b = blockIdx.x, t = threadIdx.x;
  for (int j = 0; j < R_; ++j) A[j][t] = G[(long)b * R_ * R_ + j * R_ + t];
  __syncthreads();
  for (int j = 0; j < R_; ++j) {
    if (t == j) {
      float s = A[j][j];
      for (int k = 0; k < j; ++k) s -= A[j][k] * A[j][k];
      A[j][j] = sqrtf(fmaxf(s, 1e-20f));
    }
    __syncthreads();
    if (t > j) {
      float s = A[t][j];
      for (int k = 0; k < j; ++k) s -= A[t][k] * A[j][k];
      A[t][j] = s / A[j][j];
    }
    __syncthreads();
  }
  for (int j = 0; j < R_; ++j)
    Lg[(long)b * R_ * R_ + t * R_ + j] = (j <= t) ? A[t][j] : 0.0f;
}

// Vt = L^{-1} Wt  (forward substitution over rows), one block per batch
__global__ void k_trisolve(const float* __restrict__ Lg,
                           const float* __restrict__ Wt,
                           float* __restrict__ Vt,
                           __bf16* __restrict__ Vtb) {
  __shared__ float L[R_][R_];
  const int b = blockIdx.x, t = threadIdx.x;
  for (int i = t; i < R_ * R_; i += blockDim.x)
    (&L[0][0])[i] = Lg[(long)b * R_ * R_ + i];
  __syncthreads();
  const long base = (long)b * R_ * D_;
  for (int i = 0; i < R_; ++i) {
    const float inv = 1.0f / L[i][i];
    for (int d = t; d < D_; d += 256) {
      float s = Wt[base + (long)i * D_ + d];
      for (int j = 0; j < i; ++j) s -= L[i][j] * Vt[base + (long)j * D_ + d];
      const float q = s * inv;
      Vt[base + (long)i * D_ + d] = q;
      Vtb[base + (long)i * D_ + d] = (__bf16)q;
    }
    __syncthreads();
  }
}

// Sign fix vs anchor; emit signed Vt (R,D) bf16 and V (D,R) bf16
__global__ void k_signfix(const float* __restrict__ Vt,
                          __bf16* __restrict__ Vtb,
                          __bf16* __restrict__ Vb) {
  const int i = blockIdx.x * blockDim.x + threadIdx.x;
  if (i >= B_ * R_ * D_) return;
  const int d = i % D_;
  const int r = (i / D_) % R_;
  const int b = i / (R_ * D_);
  const float s0 = Vt[((long)b * R_ + r) * D_];    // V[0][r] = dot with anchor
  const float sg = (s0 + 1e-12f) >= 0.0f ? 1.0f : -1.0f;
  const float v = Vt[i] * sg;
  Vtb[i] = (__bf16)v;
  Vb[((long)b * D_ + d) * R_ + r] = (__bf16)v;
}

// traces = xp @ V (row-major f32). Same structure as k_gemm_Yt.
__global__ void k_gemm_traces(const __bf16* __restrict__ xpb,
                              const __bf16* __restrict__ Vtb,
                              float* __restrict__ tr) {
  __shared__ __bf16 Bs[64][LDP_];
  const int tid = threadIdx.x;
  const int wave = blockIdx.x * 8 + (tid >> 5);
  const int lane = tid & 31;
  const int b = wave >> 7;
  const int tm = wave & 127;
  const __bf16* A  = xpb + (long)b * T_ * D_;
  const __bf16* Bp = Vtb + (long)b * R_ * D_;
  const int ar = tm * 16 + (lane & 15);
  v8f acc[4] = {};
  for (int kc = 0; kc < D_; kc += KC_) {
    __syncthreads();
    for (int idx = tid * 8; idx < 64 * KC_; idx += 2048) {
      const int rr = idx >> 8, kk = idx & (KC_ - 1);
      *(v8bf*)&Bs[rr][kk] = *(const v8bf*)(Bp + (long)rr * D_ + kc + kk);
    }
    __syncthreads();
    for (int k0 = 0; k0 < KC_; k0 += 32) {
      v16bf a = frag_row(A, ar, D_, kc + k0, lane);
      v16bf bf[4];
#pragma unroll
      for (int j = 0; j < 4; ++j)
        bf[j] = frag_s(&Bs[j * 16 + (lane & 15)][0], k0, lane);
#pragma unroll
      for (int j = 0; j < 4; ++j) acc[j] = WMMA_BF16(a, bf[j], acc[j]);
    }
  }
  float* O = tr + (long)b * T_ * R_;
  const int rbase = tm * 16 + ((lane >> 4) << 3);
#pragma unroll
  for (int j = 0; j < 4; ++j) {
    const int col = j * 16 + (lane & 15);
#pragma unroll
    for (int r = 0; r < 8; ++r)
      O[(long)(rbase + r) * R_ + col] = acc[j][r];
  }
}

// scales[b][r] = sqrt(sum_t traces^2 + EPS); one block per (b,r)
__global__ void k_scales(const float* __restrict__ tr,
                         float* __restrict__ sc) {
  __shared__ float red[256];
  const int br = blockIdx.x;
  const int b = br / R_, r = br % R_;
  const float* p = tr + (long)b * T_ * R_ + r;
  float s = 0.0f;
  for (int t = threadIdx.x; t < T_; t += 256) {
    const float v = p[(long)t * R_];
    s += v * v;
  }
  red[threadIdx.x] = s;
  __syncthreads();
  for (int w = 128; w > 0; w >>= 1) {
    if (threadIdx.x < w) red[threadIdx.x] += red[threadIdx.x + w];
    __syncthreads();
  }
  if (threadIdx.x == 0) sc[br] = sqrtf(red[0] + EPSF);
}

// tn = shrink(traces/scales); two spiral flows. One thread per (b,t).
__global__ void k_shrink_spiral(const float* __restrict__ tr,
                                const float* __restrict__ sc,
                                float* __restrict__ tp) {
  const int bt = blockIdx.x * blockDim.x + threadIdx.x;
  if (bt >= BT_) return;
  const int b = bt / T_;
  const float* row = tr + (long)bt * R_;
  const float* s = sc + b * R_;
  float v[R_];
#pragma unroll
  for (int r = 0; r < R_; ++r) {
    const float tn = row[r] / s[r];
    const float g = gelu_exact(fabsf(tn) - LAMF);
    v[r] = (tn >= 0.0f) ? g : -g;
  }
  const float c = 0.99500416527803f, sn = 0.09983341664683f;  // cos/sin(0.1)
#pragma unroll
  for (int it = 0; it < 2; ++it) {
    float dot = 0.0f;
#pragma unroll
    for (int r = 0; r < R_; ++r) dot += v[r] * v[r];
    const float nrm = fmaxf(sqrtf(dot), 1e-8f);
    const float rc = (6.0f - nrm) / nrm;
#pragma unroll
    for (int p = 0; p < R_ / 2; ++p) {
      const float xi = v[2 * p], xj = v[2 * p + 1];
      const float ri = c * xi - sn * xj;
      const float rj = sn * xi + c * xj;
      v[2 * p]     = xi + 0.1f * (ri - xi + rc * xi);
      v[2 * p + 1] = xj + 0.1f * (rj - xj + rc * xj);
    }
  }
  float* o = tp + (long)bt * R_;
#pragma unroll
  for (int r = 0; r < R_; ++r) o[r] = v[r];
}

// AR(1) scan along T, fold in *scales, emit bf16 u. One thread per (b,r).
__global__ void k_ar1_u(const float* __restrict__ tp,
                        const float* __restrict__ sc,
                        __bf16* __restrict__ u) {
  const int i = blockIdx.x * blockDim.x + threadIdx.x;
  if (i >= B_ * R_) return;
  const int b = i / R_, r = i % R_;
  const float s = sc[i];
  const float* p = tp + (long)b * T_ * R_ + r;
  __bf16* o = u + (long)b * T_ * R_ + r;
  float y = p[0];
  o[0] = (__bf16)(y * s);
  for (int t = 1; t < T_; ++t) {
    y = RHOF * y + (1.0f - RHOF) * p[(long)t * R_];
    o[(long)t * R_] = (__bf16)(y * s);
  }
}

// x_tilde = u @ V^T ; x_hat = x_tilde - xp + x.  M=BT, N=D, K=R.
// 32x64 per wave; block shares one 64x64 V slab (staged once).
__global__ void k_gemm_xh(const __bf16* __restrict__ u,
                          const __bf16* __restrict__ Vb,
                          const float* __restrict__ xp,
                          const float* __restrict__ x,
                          __bf16* __restrict__ xhb) {
  __shared__ __bf16 Bs[64][R_ + 8];
  const int tid = threadIdx.x;
  const int wave = blockIdx.x * 8 + (tid >> 5);
  const int lane = tid & 31;
  const int bn = wave >> 8, bm = wave & 255;  // block shares bn and batch
  const int m0 = bm * 32, n0 = bn * 64;
  const int b = m0 >> 11;
  const __bf16* Bp = Vb + (long)b * D_ * R_;
  for (int idx = tid * 8; idx < 64 * R_; idx += 2048) {
    const int rr = idx >> 6, kk = idx & 63;
    *(v8bf*)&Bs[rr][kk] = *(const v8bf*)(Bp + (long)(n0 + rr) * R_ + kk);
  }
  __syncthreads();
  v8f acc[2][4] = {};
#pragma unroll
  for (int k0 = 0; k0 < R_; k0 += 32) {
    v16bf a[2], bf[4];
#pragma unroll
    for (int i = 0; i < 2; ++i)
      a[i] = frag_row(u, m0 + i * 16 + (lane & 15), R_, k0, lane);
#pragma unroll
    for (int j = 0; j < 4; ++j)
      bf[j] = frag_s(&Bs[j * 16 + (lane & 15)][0], k0, lane);
#pragma unroll
    for (int i = 0; i < 2; ++i)
#pragma unroll
      for (int j = 0; j < 4; ++j)
        acc[i][j] = WMMA_BF16(a[i], bf[j], acc[i][j]);
  }
#pragma unroll
  for (int i = 0; i < 2; ++i) {
    const int rbase = m0 + i * 16 + ((lane >> 4) << 3);
#pragma unroll
    for (int j = 0; j < 4; ++j) {
      const int col = n0 + j * 16 + (lane & 15);
#pragma unroll
      for (int r = 0; r < 8; ++r) {
        const long idx = (long)(rbase + r) * D_ + col;
        xhb[idx] = (__bf16)(acc[i][j][r] - xp[idx] + x[idx]);
      }
    }
  }
}

// y = x_hat @ w_out^T ; out = delta*tanh(scale*y)+bias. M=BT, N=D, K=D.
// 32x64 per wave; block-shared 64-col w_out slab staged in LDS chunks.
__global__ void k_gemm_out(const __bf16* __restrict__ xhb,
                           const __bf16* __restrict__ wo,
                           const float* __restrict__ tscale,
                           const float* __restrict__ tdelta,
                           const float* __restrict__ tbias,
                           float* __restrict__ out) {
  __shared__ __bf16 Bs[64][LDP_];
  const int tid = threadIdx.x;
  const int wave = blockIdx.x * 8 + (tid >> 5);
  const int lane = tid & 31;
  const int bn = wave >> 8, bm = wave & 255;  // block shares bn
  const int m0 = bm * 32, n0 = bn * 64;
  v8f acc[2][4] = {};
  for (int kc = 0; kc < D_; kc += KC_) {
    __syncthreads();
    for (int idx = tid * 8; idx < 64 * KC_; idx += 2048) {
      const int rr = idx >> 8, kk = idx & (KC_ - 1);
      *(v8bf*)&Bs[rr][kk] = *(const v8bf*)(wo + (long)(n0 + rr) * D_ + kc + kk);
    }
    __syncthreads();
    for (int k0 = 0; k0 < KC_; k0 += 32) {
      __builtin_prefetch(xhb + (long)(m0 + (lane & 15)) * D_ + kc + k0 + 128, 0, 1);
      v16bf a[2], bf[4];
#pragma unroll
      for (int i = 0; i < 2; ++i)
        a[i] = frag_row(xhb, m0 + i * 16 + (lane & 15), D_, kc + k0, lane);
#pragma unroll
      for (int j = 0; j < 4; ++j)
        bf[j] = frag_s(&Bs[j * 16 + (lane & 15)][0], k0, lane);
#pragma unroll
      for (int i = 0; i < 2; ++i)
#pragma unroll
        for (int j = 0; j < 4; ++j)
          acc[i][j] = WMMA_BF16(a[i], bf[j], acc[i][j]);
    }
  }
  const float ts = tscale[0];
#pragma unroll
  for (int i = 0; i < 2; ++i) {
    const int rbase = m0 + i * 16 + ((lane >> 4) << 3);
#pragma unroll
    for (int j = 0; j < 4; ++j) {
      const int col = n0 + j * 16 + (lane & 15);
      const float dl = tdelta[col], bi = tbias[col];
#pragma unroll
      for (int r = 0; r < 8; ++r) {
        const long idx = (long)(rbase + r) * D_ + col;
        out[idx] = dl * tanhf(ts * acc[i][j][r]) + bi;
      }
    }
  }
}

// ---------------------------------------------------------------------------
extern "C" void kernel_launch(void* const* d_in, const int* in_sizes, int n_in,
                              void* d_out, int out_size, void* d_ws, size_t ws_size,
                              hipStream_t stream) {
  const float* x        = (const float*)d_in[0];
  const float* w_in     = (const float*)d_in[1];
  const float* w_out_s  = (const float*)d_in[2];
  const float* b_s      = (const float*)d_in[3];
  const float* w_out    = (const float*)d_in[4];
  const float* tn_scale = (const float*)d_in[5];
  const float* tn_delta = (const float*)d_in[6];
  const float* tn_bias  = (const float*)d_in[7];
  float* out = (float*)d_out;

  char* wp = (char*)d_ws;
  auto alloc = [&](size_t bytes) -> char* {
    char* p = wp;
    wp += (bytes + 255) & ~(size_t)255;
    return p;
  };
  __bf16* xb    = (__bf16*)alloc((size_t)BT_ * D_ * 2);   // reused as xh_bf
  __bf16* wib   = (__bf16*)alloc((size_t)R_ * D_ * 2);
  __bf16* wosb  = (__bf16*)alloc((size_t)D_ * R_ * 2);
  __bf16* wob   = (__bf16*)alloc((size_t)D_ * D_ * 2);
  __bf16* zb    = (__bf16*)alloc((size_t)BT_ * R_ * 2);
  float*  xp    = (float*) alloc((size_t)BT_ * D_ * 4);
  __bf16* xpb   = (__bf16*)alloc((size_t)BT_ * D_ * 2);
  __bf16* xpT   = (__bf16*)alloc((size_t)BT_ * D_ * 2);   // (B, D, T)
  float*  Vt    = (float*) alloc((size_t)B_ * R_ * D_ * 4);
  __bf16* Vtb   = (__bf16*)alloc((size_t)B_ * R_ * D_ * 2);
  __bf16* Vb    = (__bf16*)alloc((size_t)B_ * D_ * R_ * 2);
  __bf16* Yt    = (__bf16*)alloc((size_t)B_ * R_ * T_ * 2);
  float*  Wt    = (float*) alloc((size_t)B_ * R_ * D_ * 4);
  __bf16* Wtb   = (__bf16*)alloc((size_t)B_ * R_ * D_ * 2);
  float*  G     = (float*) alloc((size_t)B_ * R_ * R_ * 4);
  float*  Lg    = (float*) alloc((size_t)B_ * R_ * R_ * 4);
  float*  tr    = (float*) alloc((size_t)B_ * T_ * R_ * 4);
  float*  sc    = (float*) alloc((size_t)B_ * R_ * 4);
  float*  tp    = (float*) alloc((size_t)B_ * T_ * R_ * 4);
  __bf16* u     = (__bf16*)alloc((size_t)B_ * T_ * R_ * 2);
  __bf16* xhb   = xb;  // x_bf is dead after k_gemm_z; reuse for x_hat bf16

  // --- Precision conversions -------------------------------------------------
  k_f32_to_bf16<<<2048, 256, 0, stream>>>(x, xb, BT_ * D_);
  k_f32_to_bf16<<<256, 256, 0, stream>>>(w_in, wib, R_ * D_);
  k_f32_to_bf16<<<256, 256, 0, stream>>>(w_out_s, wosb, D_ * R_);
  k_f32_to_bf16<<<1024, 256, 0, stream>>>(w_out, wob, D_ * D_);

  // --- Shift path + centered input ------------------------------------------
  k_gemm_z<<<64, 256, 0, stream>>>(xb, wib, zb);
  k_gemm_s_xp<<<512, 256, 0, stream>>>(zb, wosb, b_s, x, xp, xpb, xpT);

  // --- Subspace iteration (4 rounds of cov_matvec + Cholesky-QR) -------------
  k_init_V<<<(B_ * R_ * D_ + 255) / 256, 256, 0, stream>>>(Vt, Vtb);
  for (int it = 0; it < 4; ++it) {
    k_gemm_Yt<<<64, 256, 0, stream>>>(xpb, Vtb, Yt);
    k_gemm_Wt<<<16, 256, 0, stream>>>(Yt, xpT, Vt, Wt, Wtb);
    k_gemm_G<<<8, 256, 0, stream>>>(Wtb, G);
    k_cholesky<<<B_, 64, 0, stream>>>(G, Lg);
    k_trisolve<<<B_, 256, 0, stream>>>(Lg, Wt, Vt, Vtb);
  }
  k_signfix<<<(B_ * R_ * D_ + 255) / 256, 256, 0, stream>>>(Vt, Vtb, Vb);

  // --- Traces, scales, shrink + spiral + AR(1) -------------------------------
  k_gemm_traces<<<64, 256, 0, stream>>>(xpb, Vtb, tr);
  k_scales<<<B_ * R_, 256, 0, stream>>>(tr, sc);
  k_shrink_spiral<<<(BT_ + 127) / 128, 128, 0, stream>>>(tr, sc, tp);
  k_ar1_u<<<1, 256, 0, stream>>>(tp, sc, u);

  // --- Reconstruction + output projection ------------------------------------
  k_gemm_xh<<<512, 256, 0, stream>>>(u, Vb, xp, x, xhb);
  k_gemm_out<<<512, 256, 0, stream>>>(xhb, wob, tn_scale, tn_delta, tn_bias, out);

  (void)in_sizes; (void)n_in; (void)out_size; (void)ws_size;
}
